// CrossAttention_33947421507772
// MI455X (gfx1250) — matrix operand order repaired and checked by reading, not compiled, not run on hardware
//
#include <hip/hip_runtime.h>
#include <hip/hip_bf16.h>

typedef _Float16 half_t;
typedef __attribute__((ext_vector_type(16))) _Float16 v16h;
typedef __attribute__((ext_vector_type(8)))  float    v8f;
typedef unsigned int u32x4 __attribute__((ext_vector_type(4)));
typedef int          i32x4 __attribute__((ext_vector_type(4)));
typedef int          i32x8 __attribute__((ext_vector_type(8)));

#define B_   16
#define LQ   128
#define LP   512
#define H_   32
#define DH   32
#define INNER 1024
#define CH   64
#define HID  128

// ---------------------------------------------------------------------------
// WMMA operand layout helpers (CDNA5 ISA, wave32)
// A (16x32 f16): lane = 16*hi + m ; halves [0..7]=K(hi*8 .. hi*8+7),
//                halves [8..15]=K(16+hi*8 .. 16+hi*8+7)
// B (32x16 f16): lane = 16*hi + n ; halves use the same K order as A
// C/D (16x16 f32, 8 VGPRs): M = r + 8*(lane>>4), N = lane & 15
// ---------------------------------------------------------------------------
__device__ __forceinline__ v16h load_A16(const half_t* base, int ldk, int lane) {
  int m = lane & 15, hi = lane >> 4;
  v16h a;
#pragma unroll
  for (int t = 0; t < 8; ++t) a[t] = base[m * ldk + hi * 8 + t];
#pragma unroll
  for (int t = 0; t < 8; ++t) a[t + 8] = base[m * ldk + 16 + hi * 8 + t];
  return a;
}

// Same but only K 0..15 valid; K 16..31 zero-padded.
__device__ __forceinline__ v16h load_A16_k16(const half_t* base, int ldk, int lane) {
  int m = lane & 15, hi = lane >> 4;
  v16h a;
#pragma unroll
  for (int t = 0; t < 8; ++t) a[t] = base[m * ldk + hi * 8 + t];
#pragma unroll
  for (int t = 0; t < 8; ++t) a[t + 8] = (half_t)0.0f;
  return a;
}

__device__ __forceinline__ v16h load_B16(const half_t* base, int ldn, int lane) {
  v16h b;
#pragma unroll
  for (int t = 0; t < 16; ++t) b[t] = base[((lane >> 4) * 8 + (t & 7) + (t >> 3) * 16) * ldn + (lane & 15)];
  return b;
}

__device__ __forceinline__ v8f wmma_f16(v16h a, v16h b, v8f c) {
  return __builtin_amdgcn_wmma_f32_16x16x32_f16(false, a, false, b, (short)0, c,
                                                false, false);
}

__device__ __forceinline__ float mish_f(float x) {
  float sp = (x > 20.0f) ? x : log1pf(__expf(x));
  return x * tanhf(sp);
}

// ---------------------------------------------------------------------------
// Generic GEMM: C[M,N] = A[M,K] (f32) * Bw[K,N] (f32) (+bias), f16 WMMA inside.
// mode 0: store f32 row-major. mode 1: store f16 row-major.
// mode 2: store f16 "cqkT" layout [b][head][d][j] (m = b*512+j, n = head*32+d).
// Block: 256 threads, 64x64 C tile. Grid: (N/64, M/64).
// ---------------------------------------------------------------------------
__global__ void __launch_bounds__(256)
wmma_gemm(const float* __restrict__ A, const float* __restrict__ Bw,
          const float* __restrict__ bias, void* Cout, int M, int N, int K,
          int mode) {
  __shared__ half_t At[64 * 64];
  __shared__ half_t Bt[64 * 64];
  const int m0 = blockIdx.y * 64, n0 = blockIdx.x * 64;
  const int tid = threadIdx.x, lane = tid & 31, wave = tid >> 5;

  v8f acc[2];
  v8f vz = {};
  acc[0] = vz; acc[1] = vz;

  for (int kp = 0; kp < K; kp += 64) {
    for (int idx = tid; idx < 64 * 64; idx += 256) {
      int r = idx >> 6, c = idx & 63;
      At[idx] = (half_t)A[(long)(m0 + r) * K + kp + c];
      Bt[idx] = (half_t)Bw[(long)(kp + r) * N + n0 + c];
    }
    __syncthreads();
#pragma unroll
    for (int s = 0; s < 2; ++s) {
      int st = wave * 2 + s;
      int mt = st >> 2, nt = st & 3;
#pragma unroll
      for (int k2 = 0; k2 < 2; ++k2) {
        v16h a = load_A16(&At[(mt * 16) * 64 + k2 * 32], 64, lane);
        v16h b = load_B16(&Bt[(k2 * 32) * 64 + nt * 16], 64, lane);
        acc[s] = wmma_f16(a, b, acc[s]);
      }
    }
    __syncthreads();
  }

#pragma unroll
  for (int s = 0; s < 2; ++s) {
    int st = wave * 2 + s;
    int mt = st >> 2, nt = st & 3;
    int n = n0 + nt * 16 + (lane & 15);
    int mb = m0 + mt * 16 + ((lane >> 4) << 3);
#pragma unroll
    for (int r = 0; r < 8; ++r) {
      int m = mb + r;
      float v = acc[s][r] + (bias ? bias[n] : 0.0f);
      if (mode == 0) {
        ((float*)Cout)[(long)m * N + n] = v;
      } else if (mode == 1) {
        ((half_t*)Cout)[(long)m * N + n] = (half_t)v;
      } else {
        int bb = m >> 9, j = m & 511;       // Lp = 512
        int head = n >> 5, d = n & 31;      // dh = 32
        ((half_t*)Cout)[((long)(bb * H_ + head) * DH + d) * LP + j] = (half_t)v;
      }
    }
  }
}

// ---------------------------------------------------------------------------
// Fused: sim -> concat(dis) -> channel MLP -> online softmax -> AV + dis-agg
// Grid: 128 blocks = (b, 16-row ligand tile). 256 threads = 8 waves.
// batch_dis_emb chunk is brought in by the Tensor Data Mover (async DMA into
// the hbuf LDS region, which is dead at chunk start), overlapped with the sim
// WMMAs, then synchronized with s_wait_tensorcnt.
// ---------------------------------------------------------------------------
#define SMEM_BYTES 134400
#define HBUF_LDS_OFF 65536u   // static LDS of this kernel starts at offset 0

__global__ void __launch_bounds__(256)
attn_fused(const half_t* __restrict__ qk, const half_t* __restrict__ cqkT,
           const half_t* __restrict__ cv, const float* __restrict__ dis_emb,
           const float* __restrict__ bdis, const float* __restrict__ W_m1,
           const float* __restrict__ W_m2, const float* __restrict__ b_m2,
           const float* __restrict__ W_d1, const float* __restrict__ b_d1,
           const float* __restrict__ W_d2, const float* __restrict__ b_d2,
           float* __restrict__ attn_out, float* __restrict__ out_dis) {
  __shared__ __align__(16) char smem[SMEM_BYTES];
  half_t* sWm1  = (half_t*)(smem);                 // [64][128]  16384 B
  half_t* sWm2  = (half_t*)(smem + 16384);         // [128][64]  16384 B
  half_t* s_mat = (half_t*)(smem + 32768);         // [256][64]  32768 B (also logits)
  half_t* hbuf  = (half_t*)(smem + 65536);         // [128][128] 32768 B (TDM stage too)
  half_t* pbuf  = (half_t*)(smem + 98304);         // [32][16][16] 16384 B
  float*  mbuf  = (float*)(smem + 114688);         // [64][16]
  float*  lbuf  = (float*)(smem + 118784);         // [64][16]
  float*  sfb   = (float*)(smem + 122880);         // [32][16]
  float*  dacc  = (float*)(smem + 124928);         // [16][32][3]
  float*  bdis_s= (float*)(smem + 131072);         // [16][16][3]
  float*  bm2s  = (float*)(smem + 134144);         // [64]
  const float* disstage = (const float*)(smem + 65536);  // TDM dest [16j][16i][32ch] f32

  const int tid = threadIdx.x, lane = tid & 31, wave = tid >> 5;
  const int blk = blockIdx.x;
  const int b = blk >> 3;
  const int i0 = (blk & 7) * 16;
  const int head0 = wave * 4;

  for (int idx = tid; idx < 64 * HID; idx += 256) sWm1[idx] = (half_t)W_m1[idx];
  for (int idx = tid; idx < HID * 64; idx += 256) sWm2[idx] = (half_t)W_m2[idx];
  for (int idx = tid; idx < 64; idx += 256) bm2s[idx] = b_m2[idx];
  for (int idx = tid; idx < 1024; idx += 256) { mbuf[idx] = -1e30f; lbuf[idx] = 0.0f; }
  for (int idx = tid; idx < 16 * 32 * 3; idx += 256) dacc[idx] = 0.0f;

  // Per-wave persistent: qk A-operands (4 heads) + AV accumulators in regs.
  v16h aqk[4];
#pragma unroll
  for (int hh = 0; hh < 4; ++hh) {
    int head = head0 + hh;
    const half_t* src = qk + ((long)(b * LQ + i0) * H_ + head) * DH;
    aqk[hh] = load_A16(src, H_ * DH, lane);  // row stride 1024 (per ligand row)
  }
  v8f vz = {};
  v8f oacc[4][2];
#pragma unroll
  for (int hh = 0; hh < 4; ++hh) { oacc[hh][0] = vz; oacc[hh][1] = vz; }

  __syncthreads();

  for (int jc = 0; jc < 32; ++jc) {
    const int j0 = jc * 16;

    // --- TDM: async 3D-tile load of batch_dis_emb chunk into LDS (hbuf area).
    // Tile: x = ch (32 contiguous f32), y = i (16, stride 32), z = j (16,
    // stride 128*32). D# per CDNA5 ISA.
    if (wave == 0) {
      unsigned long long ga = (unsigned long long)(const void*)(
          dis_emb + ((long)(b * LP + j0) * LQ + i0) * H_);
      u32x4 g0;
      g0[0] = 1u;                                    // count=1, user descriptor
      g0[1] = HBUF_LDS_OFF;                          // lds_addr (bytes)
      g0[2] = (unsigned)(ga & 0xffffffffu);          // global_addr[31:0]
      g0[3] = (unsigned)((ga >> 32) & 0x01ffffffu)   // global_addr[56:32]
              | (2u << 30);                          // type = 2 ("image")
      i32x8 g1;
      g1[0] = (int)(2u << 16);                       // data_size = 4 bytes
      g1[1] = (int)(32u << 16);                      // tensor_dim0 = 32 (lo16)
      g1[2] = (int)(128u << 16);                     // dim0 hi | tensor_dim1 = 128
      g1[3] = (int)(32u << 16);                      // dim1 hi | tile_dim0 = 32
      g1[4] = (int)(16u | (16u << 16));              // tile_dim1=16, tile_dim2=16
      g1[5] = 32;                                    // tensor_dim0_stride = 32
      g1[6] = (int)(4096u << 16);                    // tensor_dim1_stride = 4096
      g1[7] = 0;
      i32x4 g2; g2[0] = 512; g2[1] = 0; g2[2] = 0; g2[3] = 0;  // tensor_dim2
      i32x4 g3; g3[0] = 0;  g3[1] = 0; g3[2] = 0; g3[3] = 0;
      i32x8 g4 = {};                                 // extra operand (clang-23
                                                     // 6-arg form), zero-filled
      __builtin_amdgcn_tensor_load_to_lds(g0, g1, g2, g3, g4, 0);
    }

    // prefetch next chunk's cv rows into cache while we compute this one
    if (jc + 1 < 32) {
      const half_t* nc =
          cv + ((long)(b * LP + j0 + 16 + (lane & 15)) * H_ + head0) * DH;
      __builtin_prefetch(nc, 0, 1);
    }

    // stage batch_dis chunk [j][i][d]
    for (int idx = tid; idx < 16 * 16 * 3; idx += 256) {
      int j = idx / 48, rem = idx % 48, i = rem / 3, d = rem % 3;
      bdis_s[idx] = bdis[((long)(b * LP + j0 + j) * LQ + i0 + i) * 3 + d];
    }

    // sim channels 0..31: one WMMA per head (overlaps the TDM transfer)
#pragma unroll
    for (int hh = 0; hh < 4; ++hh) {
      int head = head0 + hh;
      const half_t* bsrc = cqkT + ((long)(b * H_ + head) * DH) * LP + j0;
      v16h bv = load_B16(bsrc, LP, lane);
      v8f c = wmma_f16(aqk[hh], bv, vz);
      int n = lane & 15, mb = (lane >> 4) << 3;
#pragma unroll
      for (int r = 0; r < 8; ++r)
        s_mat[(n * 16 + mb + r) * 64 + head] = (half_t)(c[r] * 0.17677669529f);
    }

    if (wave == 0) __builtin_amdgcn_s_wait_tensorcnt(0);
    __syncthreads();

    // dis-emb channels 32..63 of s matrix (row = j*16 + i), from TDM-staged LDS
    {
      int row = tid;  // 256 threads == 256 rows
#pragma unroll
      for (int c = 0; c < 32; ++c)
        s_mat[row * 64 + 32 + c] = (half_t)disstage[row * 32 + c];
    }
    __syncthreads();

    // channel MLP: two 128-row halves through GEMM1(64->128)+mish, GEMM2(128->64)
    for (int hf = 0; hf < 2; ++hf) {
      const int rbase = hf * 128;
      for (int t = wave; t < 64; t += 8) {  // 8 mtiles x 8 ntiles
        int mt = t >> 3, nt = t & 7;
        v8f c = vz;
#pragma unroll
        for (int k2 = 0; k2 < 2; ++k2) {
          v16h a = load_A16(&s_mat[(rbase + mt * 16) * 64 + k2 * 32], 64, lane);
          v16h bb = load_B16(&sWm1[(k2 * 32) * HID + nt * 16], HID, lane);
          c = wmma_f16(a, bb, c);
        }
        int n = lane & 15, mb = (lane >> 4) << 3;
#pragma unroll
        for (int r = 0; r < 8; ++r)
          hbuf[(mt * 16 + mb + r) * HID + nt * 16 + n] = (half_t)mish_f(c[r]);
      }
      __syncthreads();
      for (int t = wave; t < 32; t += 8) {  // 8 mtiles x 4 ntiles
        int mt = t >> 2, nt = t & 3;
        v8f c = vz;
#pragma unroll
        for (int k4 = 0; k4 < 4; ++k4) {
          v16h a = load_A16(&hbuf[(mt * 16) * HID + k4 * 32], HID, lane);
          v16h bb = load_B16(&sWm2[(k4 * 32) * CH + nt * 16], CH, lane);
          c = wmma_f16(a, bb, c);
        }
        int n = lane & 15, mb = (lane >> 4) << 3;
#pragma unroll
        for (int r = 0; r < 8; ++r)
          s_mat[(rbase + mt * 16 + mb + r) * 64 + nt * 16 + n] =
              (half_t)(c[r] + bm2s[nt * 16 + n]);
      }
      __syncthreads();
    }

    // online softmax over j per (i, channel); dis channels aggregated here
    for (int idx = tid; idx < 1024; idx += 256) {
      int ch = idx & 63, i = idx >> 6;
      float mold = mbuf[ch * 16 + i];
      float vmax = mold;
      float p[16];
#pragma unroll
      for (int j = 0; j < 16; ++j) {
        float v = (float)s_mat[(j * 16 + i) * 64 + ch];
        p[j] = v;
        vmax = fmaxf(vmax, v);
      }
      float scale = __expf(mold - vmax);
      float lsum = lbuf[ch * 16 + i] * scale;
#pragma unroll
      for (int j = 0; j < 16; ++j) { p[j] = __expf(p[j] - vmax); lsum += p[j]; }
      mbuf[ch * 16 + i] = vmax;
      lbuf[ch * 16 + i] = lsum;
      if (ch < 32) {
        sfb[ch * 16 + i] = scale;
#pragma unroll
        for (int j = 0; j < 16; ++j) pbuf[(ch * 16 + i) * 16 + j] = (half_t)p[j];
      } else {
        int c2 = ch - 32;
#pragma unroll
        for (int d = 0; d < 3; ++d) {
          float a = dacc[(i * 32 + c2) * 3 + d] * scale;
#pragma unroll
          for (int j = 0; j < 16; ++j) a += p[j] * bdis_s[(j * 16 + i) * 3 + d];
          dacc[(i * 32 + c2) * 3 + d] = a;
        }
      }
    }
    __syncthreads();

    // AV: P[16i x 16j] (zero-padded K=32) x cv[32j x 32d], accumulators in regs
#pragma unroll
    for (int hh = 0; hh < 4; ++hh) {
      int head = head0 + hh;
      v16h a = load_A16_k16(&pbuf[(head * 16) * 16], 16, lane);
      int mb = (lane >> 4) << 3;
      int jj = j0 + (lane >> 4) * 8;  // lane = 16*hi + n; elements 0..7 = j jj..jj+7
      const half_t* bbase = cv + ((long)(b * LP + jj) * H_ + head) * DH + (lane & 15);
#pragma unroll
      for (int nt = 0; nt < 2; ++nt) {
#pragma unroll
        for (int r = 0; r < 8; ++r) oacc[hh][nt][r] *= sfb[head * 16 + mb + r];
        v16h bv;
#pragma unroll
        for (int t = 0; t < 16; ++t) bv[t] = t < 8 ? bbase[(long)t * INNER + nt * 16] : (half_t)0.0f;
        oacc[hh][nt] = wmma_f16(a, bv, oacc[hh][nt]);
      }
    }
    __syncthreads();
  }

  // epilogue: normalized attention output -> workspace (for @W_o GEMM)
#pragma unroll
  for (int hh = 0; hh < 4; ++hh) {
    int head = head0 + hh;
    int n = lane & 15, mb = (lane >> 4) << 3;
#pragma unroll
    for (int nt = 0; nt < 2; ++nt) {
#pragma unroll
      for (int r = 0; r < 8; ++r) {
        int i = mb + r;
        attn_out[(long)(b * LQ + i0 + i) * INNER + head * DH + nt * 16 + n] =
            oacc[hh][nt][r] / lbuf[head * 16 + i];
      }
    }
  }
  __syncthreads();

  // normalize dis accumulators, then tiny dis-MLP (48 rows of (i,d))
  for (int idx = tid; idx < 16 * 32 * 3; idx += 256) {
    int i = idx / 96, c2 = (idx % 96) / 3;
    dacc[idx] = dacc[idx] / lbuf[(32 + c2) * 16 + i];
  }
  __syncthreads();
  float* hid = (float*)(smem + 32768);  // reuse s_mat region: 48*64 f32
  for (int idx = tid; idx < 48 * 64; idx += 256) {
    int row = idx >> 6, h2 = idx & 63;
    int i = row / 3, d = row % 3;
    float a = b_d1[h2];
    for (int c = 0; c < 32; ++c) a += dacc[(i * 32 + c) * 3 + d] * W_d1[c * 64 + h2];
    hid[row * 64 + h2] = mish_f(a);
  }
  __syncthreads();
  for (int idx = tid; idx < 48 * 32; idx += 256) {
    int row = idx >> 5, oc = idx & 31;
    int i = row / 3, d = row % 3;
    float a = b_d2[oc];
    for (int k = 0; k < 64; ++k) a += hid[row * 64 + k] * W_d2[k * 32 + oc];
    out_dis[((long)(b * LQ + i0 + i) * 3 + d) * 32 + oc] = a;
  }
}

// ---------------------------------------------------------------------------
extern "C" void kernel_launch(void* const* d_in, const int* in_sizes, int n_in,
                              void* d_out, int out_size, void* d_ws,
                              size_t ws_size, hipStream_t stream) {
  const float* x             = (const float*)d_in[0];
  const float* context       = (const float*)d_in[1];
  const float* batch_dis     = (const float*)d_in[2];
  const float* batch_dis_emb = (const float*)d_in[3];
  const float* W_qk  = (const float*)d_in[6];
  const float* W_cqk = (const float*)d_in[8];
  const float* W_cv  = (const float*)d_in[9];
  const float* W_m1  = (const float*)d_in[10];
  const float* W_m2  = (const float*)d_in[11];
  const float* b_m2  = (const float*)d_in[12];
  const float* W_o   = (const float*)d_in[13];
  const float* b_o   = (const float*)d_in[14];
  const float* W_d1  = (const float*)d_in[15];
  const float* b_d1  = (const float*)d_in[16];
  const float* W_d2  = (const float*)d_in[17];
  const float* b_d2  = (const float*)d_in[18];

  char* ws = (char*)d_ws;
  half_t* qk_h     = (half_t*)ws;                    //  4 MB  [B*Lq][INNER] f16
  half_t* cqkT     = (half_t*)(ws + (4l << 20));     // 16 MB  [B][h][dh][Lp] f16
  half_t* cv_h     = (half_t*)(ws + (20l << 20));    // 16 MB  [B*Lp][INNER] f16
  float*  attn_out = (float*)(ws + (36l << 20));     //  8 MB  [B*Lq][INNER] f32

  float* out     = (float*)d_out;
  float* out_dis = out + (long)(B_ * LQ) * 256;

  // projections (f16 outputs; cqk stored transposed for WMMA B-operands)
  wmma_gemm<<<dim3(16, 32), 256, 0, stream>>>(x, W_qk, nullptr, qk_h,
                                              B_ * LQ, INNER, 256, 1);
  wmma_gemm<<<dim3(16, 128), 256, 0, stream>>>(context, W_cqk, nullptr, cqkT,
                                               B_ * LP, INNER, 256, 2);
  wmma_gemm<<<dim3(16, 128), 256, 0, stream>>>(context, W_cv, nullptr, cv_h,
                                               B_ * LP, INNER, 256, 1);
  // fused attention core
  attn_fused<<<128, 256, 0, stream>>>(qk_h, cqkT, cv_h, batch_dis_emb, batch_dis,
                                      W_m1, W_m2, b_m2, W_d1, b_d1, W_d2, b_d2,
                                      attn_out, out_dis);
  // output projection
  wmma_gemm<<<dim3(4, 32), 256, 0, stream>>>(attn_out, W_o, b_o, out,
                                             B_ * LQ, 256, INNER, 0);
}
